// SFFBlock_54778012893786
// MI455X (gfx1250) — matrix operand order repaired and checked by reading, not compiled, not run on hardware
//
#include <hip/hip_runtime.h>
#include <hip/hip_bf16.h>
#include <math.h>

// ---------------------------------------------------------------------------
// CDNA5 (gfx1250) wave32 WMMA types
// ---------------------------------------------------------------------------
typedef __bf16 bf16_t;
typedef __attribute__((ext_vector_type(16))) __bf16 v16bf;
typedef __attribute__((ext_vector_type(8)))  float  v8f;

#define EPS 1e-5f

// Tensor geometry (fixed by the reference harness)
#define BATCH 4
#define DIM   48
#define HH    256
#define WW    256
#define HWSZ  (HH * WW)          // 65536
#define NPIX  (BATCH * HWSZ)     // 262144 total pixels

// ---------------------------------------------------------------------------
// Pre-swizzled fragment layout (CDNA5 ISA §7.12.2, wave32):
// A 16x32 bf16 fragment, element (row m, klocal):
//   lane = m + 16*((klocal>>3)&1)
//   pos  = (klocal&7) + 8*((klocal>>4)&1)
// B 32x16 bf16 fragment, element (klocal, col n):
//   lane = n + 16*((klocal>>3)&1)
//   pos  = (klocal&7) + 8*((klocal>>4)&1)
// Fragments are stored as [tile][lane][16 bf16] so each lane issues one
// contiguous 32-byte LDS read (ds_load_b128 x2) -- no v_mov_b16 repacking.
// ---------------------------------------------------------------------------
__device__ __forceinline__ int a_frag_slot(int row, int klocal) {
    int lane = row + (((klocal >> 3) & 1) << 4);
    int pos  = (klocal & 7) + (((klocal >> 4) & 1) << 3);
    return lane * 16 + pos;
}
__device__ __forceinline__ int b_frag_slot(int klocal, int ncol) {
    int lane = ncol + (((klocal >> 3) & 1) << 4);
    int pos  = (klocal & 7) + (((klocal >> 4) & 1) << 3);
    return lane * 16 + pos;
}

// ---------------------------------------------------------------------------
// K0: build per-channel 8x8 spatial kernels from ffn_fft (real rfft2 filter).
// irfft2(H * g) == h (circ-conv) w with w = irfft2 of the Hermitian extension
// of g (real & even), so precompute w once: w[c][y][x].
// ---------------------------------------------------------------------------
__global__ __launch_bounds__(256) void build_ffn_filter_kernel(
    const float* __restrict__ filt /* 48 x 8 x 5 */, float* __restrict__ wker /* 48 x 64 */) {
    int idx = blockIdx.x * 256 + threadIdx.x;
    if (idx >= DIM * 64) return;
    int c = idx >> 6, y = (idx >> 3) & 7, x = idx & 7;
    float s = 0.0f;
    for (int u = 0; u < 8; ++u) {
        for (int v = 0; v < 8; ++v) {
            float g = (v <= 4) ? filt[(c * 8 + u) * 5 + v]
                               : filt[(c * 8 + ((8 - u) & 7)) * 5 + (8 - v)];
            int ph = (u * y + v * x) & 7;
            s += g * __cosf(0.78539816339744831f * (float)ph);  // cos(2*pi*ph/8)
        }
    }
    wker[idx] = s * (1.0f / 64.0f);
}

// ---------------------------------------------------------------------------
// K1: positional 7x7 depthwise conv (pad 3) + bias + residual.  x1 = x + dw(x)
// ---------------------------------------------------------------------------
__global__ __launch_bounds__(256) void posconv_kernel(
    const float* __restrict__ x, const float* __restrict__ w,
    const float* __restrict__ bias, float* __restrict__ out) {
    long long idx = (long long)blockIdx.x * 256 + threadIdx.x;
    if (idx >= (long long)BATCH * DIM * HWSZ) return;
    int xc = (int)(idx % WW);
    int y  = (int)((idx / WW) % HH);
    int c  = (int)((idx / HWSZ) % DIM);
    int b  = (int)(idx / ((long long)DIM * HWSZ));
    const float* wc = w + c * 49;
    const float* xp = x + ((long long)(b * DIM + c)) * HWSZ;
    float s = bias[c];
    for (int dy = 0; dy < 7; ++dy) {
        int yy = y + dy - 3;
        if (yy < 0 || yy >= HH) continue;
#pragma unroll
        for (int dx = 0; dx < 7; ++dx) {
            int xx = xc + dx - 3;
            if (xx < 0 || xx >= WW) continue;
            s += xp[yy * WW + xx] * wc[dy * 7 + dx];
        }
    }
    out[idx] = x[idx] + s;
}

// ---------------------------------------------------------------------------
// Generic 1x1-conv GEMM with WMMA bf16 (f32 accumulate), fragment-swizzled LDS.
//   M tile = 64 pixels per block (HW % 64 == 0 so tiles never cross a batch)
//   AsF: [4 mtiles][KP/32][32 lanes][16] bf16   (input, optional BN fused)
//   BsF: [KP/32][COUT/16][32 lanes][16] bf16    (weights, transposed + padded)
//   8 waves cover the 4 x COUT/16 output sub-tiles; K loop fully unrolled.
// ---------------------------------------------------------------------------
template <bool BN_IN, bool IN_BF16, bool OUT_BF16, bool HAS_BIAS, int KP, int COUT>
__global__ __launch_bounds__(256) void conv1x1_wmma_kernel(
    const void* __restrict__ inp,
    const float* __restrict__ bn_w, const float* __restrict__ bn_b,
    const float* __restrict__ bn_m, const float* __restrict__ bn_v,
    const float* __restrict__ Wt /* COUT x Cin */, const float* __restrict__ bias,
    void* __restrict__ outp, int Cin) {
    constexpr int NKB    = KP / 32;
    constexpr int NTILES = COUT / 16;
    extern __shared__ char smem_raw[];
    v16bf* AsF = (v16bf*)smem_raw;                       // 4*NKB*32 fragments
    v16bf* BsF = (v16bf*)(smem_raw + (size_t)64 * KP * sizeof(bf16_t));
    bf16_t* AsE = (bf16_t*)AsF;
    bf16_t* BsE = (bf16_t*)BsF;

    const int tid = threadIdx.x;
    const long long tileBase = (long long)blockIdx.x * 64;
    const int b  = (int)(tileBase / HWSZ);
    const int p0 = (int)(tileBase % HWSZ);

    // Hint the weight stream into cache early (global_prefetch path).
    __builtin_prefetch(&Wt[(size_t)tid * 4], 0, 1);

    // ---- stage A (64 pixels x KP), coalesced 64-pixel rows per channel ----
    for (int idx = tid; idx < 64 * KP; idx += 256) {
        int c = idx >> 6, pl = idx & 63;
        float xv = 0.0f;
        if (c < Cin) {
            if (IN_BF16) xv = (float)((const bf16_t*)inp)[((long long)(b * Cin + c)) * HWSZ + p0 + pl];
            else         xv = ((const float*)inp)[((long long)(b * Cin + c)) * HWSZ + p0 + pl];
            if (BN_IN) {
                float scale = bn_w[c] * rsqrtf(bn_v[c] + EPS);
                xv = (xv - bn_m[c]) * scale + bn_b[c];
            }
        }
        int mt = pl >> 4, row = pl & 15;
        int kb = c >> 5, kl = c & 31;
        AsE[((mt * NKB + kb) * 32) * 16 + a_frag_slot(row, kl)] = (bf16_t)xv;
    }
    // ---- stage B transposed: element (k, n) = W[n][k], zero pad k >= Cin ----
    for (int idx = tid; idx < KP * COUT; idx += 256) {
        int k = idx / COUT, n = idx - k * COUT;
        int kb = k >> 5, kl = k & 31;
        int nt = n >> 4, nc = n & 15;
        BsE[((kb * NTILES + nt) * 32) * 16 + b_frag_slot(kl, nc)] =
            (bf16_t)((k < Cin) ? Wt[(size_t)n * Cin + k] : 0.0f);
    }
    __syncthreads();

    const int wave = tid >> 5, lane = tid & 31;
    for (int t = wave; t < 4 * NTILES; t += 8) {
        int mt = t / NTILES, nt = t - mt * NTILES;
        v8f acc = {};
#pragma unroll
        for (int kb = 0; kb < NKB; ++kb) {
            v16bf af = AsF[(mt * NKB + kb) * 32 + lane];
            v16bf bf = BsF[(kb * NTILES + nt) * 32 + lane];
            acc = __builtin_amdgcn_wmma_f32_16x16x32_bf16(
                false, af, false, bf, (short)0, acc, false, false);
        }
        int n  = nt * 16 + (lane & 15);
        int mb = mt * 16 + ((lane & 16) ? 8 : 0);
        float bv = HAS_BIAS ? bias[n] : 0.0f;
#pragma unroll
        for (int r = 0; r < 8; ++r) {
            long long o = ((long long)(b * COUT + n)) * HWSZ + p0 + mb + r;
            float val = acc[r] + bv;
            if (OUT_BF16) ((bf16_t*)outp)[o] = (bf16_t)val;   // 8 px merge -> b128
            else          ((float*)outp)[o]  = val;
        }
    }
}

// ---------------------------------------------------------------------------
// K3: 3x3 depthwise conv, bf16 in/out, no bias (EFA branch, 288 channels).
// ---------------------------------------------------------------------------
__global__ __launch_bounds__(256) void dwconv3_bf16_kernel(
    const bf16_t* __restrict__ in, const float* __restrict__ w,
    bf16_t* __restrict__ out, int C) {
    long long idx = (long long)blockIdx.x * 256 + threadIdx.x;
    if (idx >= (long long)BATCH * C * HWSZ) return;
    int xc = (int)(idx % WW);
    int y  = (int)((idx / WW) % HH);
    int c  = (int)((idx / HWSZ) % C);
    int b  = (int)(idx / ((long long)C * HWSZ));
    const float*  wc = w + c * 9;
    const bf16_t* ip = in + ((long long)(b * C + c)) * HWSZ;
    float s = 0.0f;
#pragma unroll
    for (int dy = 0; dy < 3; ++dy) {
        int yy = y + dy - 1;
        if (yy < 0 || yy >= HH) continue;
#pragma unroll
        for (int dx = 0; dx < 3; ++dx) {
            int xx = xc + dx - 1;
            if (xx < 0 || xx >= WW) continue;
            s += (float)ip[yy * WW + xx] * wc[dy * 3 + dx];
        }
    }
    out[idx] = (bf16_t)s;
}

// ---------------------------------------------------------------------------
// K6: 3x3 depthwise conv (384ch) + bias, then GELU gate: out = gelu(h1)*h2
// (192 output channels).  Each thread computes the matching channel pair.
// ---------------------------------------------------------------------------
__global__ __launch_bounds__(256) void dwconv3_gate_kernel(
    const bf16_t* __restrict__ in /* 2*C2 ch */, const float* __restrict__ w,
    const float* __restrict__ bias, bf16_t* __restrict__ out /* C2 ch */, int C2) {
    long long idx = (long long)blockIdx.x * 256 + threadIdx.x;
    if (idx >= (long long)BATCH * C2 * HWSZ) return;
    int xc = (int)(idx % WW);
    int y  = (int)((idx / WW) % HH);
    int c  = (int)((idx / HWSZ) % C2);
    int b  = (int)(idx / ((long long)C2 * HWSZ));
    const int Cfull = 2 * C2;
    float s1 = bias[c], s2 = bias[c + C2];
    const float*  w1 = w + c * 9;
    const float*  w2 = w + (c + C2) * 9;
    const bf16_t* i1 = in + ((long long)(b * Cfull + c)) * HWSZ;
    const bf16_t* i2 = in + ((long long)(b * Cfull + c + C2)) * HWSZ;
#pragma unroll
    for (int dy = 0; dy < 3; ++dy) {
        int yy = y + dy - 1;
        if (yy < 0 || yy >= HH) continue;
#pragma unroll
        for (int dx = 0; dx < 3; ++dx) {
            int xx = xc + dx - 1;
            if (xx < 0 || xx >= WW) continue;
            s1 += (float)i1[yy * WW + xx] * w1[dy * 3 + dx];
            s2 += (float)i2[yy * WW + xx] * w2[dy * 3 + dx];
        }
    }
    float g = 0.5f * s1 * (1.0f + erff(s1 * 0.70710678118654752f));  // exact GELU
    out[idx] = (bf16_t)(g * s2);
}

// ---------------------------------------------------------------------------
// K4: EFA core.  One block per 8x8 patch (4096 blocks).
//  o = q (8x8 circular conv) k per channel  [== irfft2(rfft2(q)*rfft2(k))]
//  channel-LayerNorm over 96 channels per pixel, m = v * LN(o),
//  then WMMA GEMM m(64x96) @ efa_out_w^T(96x48), + residual x1 -> x2.
// ---------------------------------------------------------------------------
__global__ __launch_bounds__(256) void efa_patch_kernel(
    const bf16_t* __restrict__ hidden /* 4 x 288 x H x W */,
    const float* __restrict__ ln_w, const float* __restrict__ ln_b,
    const float* __restrict__ outW /* 48 x 96 */,
    const float* __restrict__ x1, float* __restrict__ x2) {
    const int PW = WW / 8;                         // 32 patches per row
    int pb = blockIdx.x;
    int b  = pb / (PW * PW);
    int pr = (pb / PW) % PW;
    int pc = pb % PW;
    int y0 = pr * 8, x0 = pc * 8;

    __shared__ bf16_t qs[96 * 64];
    __shared__ bf16_t ks[96 * 64];
    __shared__ bf16_t vs[96 * 64];
    __shared__ v16bf  msF[4 * 3 * 32];   // GEMM A fragments [mt][kb][lane]
    __shared__ v16bf  BsF[3 * 3 * 32];   // GEMM B fragments [kb][nt][lane]
    __shared__ float  mu[64], rstd[64];
    bf16_t* msE = (bf16_t*)msF;
    bf16_t* BsE = (bf16_t*)BsF;

    const int tid = threadIdx.x;

    // stage B fragments (small, do it up front)
    for (int idx = tid; idx < 96 * 48; idx += 256) {
        int k = idx / 48, n = idx - k * 48;
        int kb = k >> 5, kl = k & 31;
        int nt = n >> 4, nc = n & 15;
        BsE[((kb * 3 + nt) * 32) * 16 + b_frag_slot(kl, nc)] = (bf16_t)outW[n * 96 + k];
    }
    // stage q / k / v patches
    for (int idx = tid; idx < 96 * 64; idx += 256) {
        int ch = idx >> 6, px = idx & 63;
        int py = px >> 3, pxx = px & 7;
        long long base = ((long long)(b * 288 + ch)) * HWSZ + (long long)(y0 + py) * WW + (x0 + pxx);
        qs[idx] = hidden[base];
        ks[idx] = hidden[base + (long long)96 * HWSZ];
        vs[idx] = hidden[base + (long long)192 * HWSZ];
    }
    __syncthreads();

    // 8x8 circular convolution per channel (24 outputs per thread)
    float oreg[24];
#pragma unroll 4
    for (int j = 0; j < 24; ++j) {
        int idx = tid + 256 * j;
        int ch = idx >> 6, px = idx & 63;
        int y = px >> 3, x = px & 7;
        const bf16_t* qc = &qs[ch * 64];
        const bf16_t* kc = &ks[ch * 64];
        float s = 0.0f;
        for (int dy = 0; dy < 8; ++dy) {
            int ry = ((y - dy) & 7) << 3;
#pragma unroll
            for (int dx = 0; dx < 8; ++dx)
                s += (float)qc[dy * 8 + dx] * (float)kc[ry | ((x - dx) & 7)];
        }
        oreg[j] = s;
    }
    __syncthreads();
    // overwrite q-buffer with o
#pragma unroll
    for (int j = 0; j < 24; ++j) qs[tid + 256 * j] = (bf16_t)oreg[j];
    __syncthreads();

    // channel LayerNorm stats per pixel (biased variance, matches reference)
    if (tid < 64) {
        float s = 0.0f, s2 = 0.0f;
        for (int ch = 0; ch < 96; ++ch) {
            float o = (float)qs[ch * 64 + tid];
            s += o; s2 += o * o;
        }
        float mean = s * (1.0f / 96.0f);
        float var  = s2 * (1.0f / 96.0f) - mean * mean;
        mu[tid] = mean;
        rstd[tid] = rsqrtf(var + EPS);
    }
    __syncthreads();

    // m = v * (LN(o)*w + b), written straight into A-fragment layout
    for (int idx = tid; idx < 96 * 64; idx += 256) {
        int ch = idx >> 6, px = idx & 63;
        float o = ((float)qs[idx] - mu[px]) * rstd[px] * ln_w[ch] + ln_b[ch];
        float mval = (float)vs[idx] * o;
        int mt = px >> 4, row = px & 15;
        int kb = ch >> 5, kl = ch & 31;
        msE[((mt * 3 + kb) * 32) * 16 + a_frag_slot(row, kl)] = (bf16_t)mval;
    }
    __syncthreads();

    // WMMA GEMM: 64x48 = (64x96)@(96x48), 12 tiles over 8 waves, K unrolled
    const int wave = tid >> 5, lane = tid & 31;
    for (int t = wave; t < 12; t += 8) {
        int mt = t / 3, nt = t - mt * 3;
        v8f acc = {};
#pragma unroll
        for (int kb = 0; kb < 3; ++kb) {
            v16bf af = msF[(mt * 3 + kb) * 32 + lane];
            v16bf bf = BsF[(kb * 3 + nt) * 32 + lane];
            acc = __builtin_amdgcn_wmma_f32_16x16x32_bf16(
                false, af, false, bf, (short)0, acc, false, false);
        }
        int n  = nt * 16 + (lane & 15);
        int mb = mt * 16 + ((lane & 16) ? 8 : 0);
#pragma unroll
        for (int r = 0; r < 8; ++r) {
            int pix = mb + r;
            long long o = ((long long)(b * DIM + n)) * HWSZ
                        + (long long)(y0 + (pix >> 3)) * WW + (x0 + (pix & 7));
            x2[o] = x1[o] + acc[r];
        }
    }
}

// ---------------------------------------------------------------------------
// K8: EDFFN spectral filter as per-channel 8x8 circular conv + residual.
// One block = one channel, 4 patches.  out = x2 + (h circ-conv w_c)
// ---------------------------------------------------------------------------
__global__ __launch_bounds__(256) void ffn_spectral_kernel(
    const float* __restrict__ h, const float* __restrict__ wker /* 48 x 64 */,
    const float* __restrict__ x2, float* __restrict__ out) {
    const int PW = WW / 8;
    const int GRP = (PW * PW) / 4;  // 256 patch-groups per channel
    int bc  = blockIdx.x;
    int grp = bc % GRP;
    int c   = (bc / GRP) % DIM;
    int b   = bc / (GRP * DIM);

    __shared__ float wk[64];
    __shared__ float hp[4][64];

    const int tid = threadIdx.x;
    if (tid < 64) wk[tid] = wker[c * 64 + tid];

    int pi = grp * 4 + (tid >> 6);
    int px = tid & 63;
    int pr = pi / PW, pc_ = pi % PW;
    int gy = pr * 8 + (px >> 3), gx = pc_ * 8 + (px & 7);
    long long base = ((long long)(b * DIM + c)) * HWSZ + (long long)gy * WW + gx;
    hp[tid >> 6][px] = h[base];
    __syncthreads();

    int y = px >> 3, x = px & 7;
    const float* hpp = hp[tid >> 6];
    float s = 0.0f;
    for (int dy = 0; dy < 8; ++dy) {
        int ry = ((y - dy) & 7) << 3;
#pragma unroll
        for (int dx = 0; dx < 8; ++dx)
            s += hpp[dy * 8 + dx] * wk[ry | ((x - dx) & 7)];
    }
    out[base] = x2[base] + s;
}

// ---------------------------------------------------------------------------
// Host launcher
// ---------------------------------------------------------------------------
extern "C" void kernel_launch(void* const* d_in, const int* in_sizes, int n_in,
                              void* d_out, int out_size, void* d_ws, size_t ws_size,
                              hipStream_t stream) {
    (void)in_sizes; (void)n_in; (void)out_size; (void)ws_size;

    const float* x         = (const float*)d_in[0];
    const float* pos_w     = (const float*)d_in[1];
    const float* pos_b     = (const float*)d_in[2];
    const float* bn1_w     = (const float*)d_in[3];
    const float* bn1_b     = (const float*)d_in[4];
    const float* bn1_m     = (const float*)d_in[5];
    const float* bn1_v     = (const float*)d_in[6];
    const float* efa_hid_w = (const float*)d_in[7];
    const float* efa_dw_w  = (const float*)d_in[8];
    const float* efa_ln_w  = (const float*)d_in[9];
    const float* efa_ln_b  = (const float*)d_in[10];
    const float* efa_out_w = (const float*)d_in[11];
    const float* bn2_w     = (const float*)d_in[12];
    const float* bn2_b     = (const float*)d_in[13];
    const float* bn2_m     = (const float*)d_in[14];
    const float* bn2_v     = (const float*)d_in[15];
    const float* ffn_in_w  = (const float*)d_in[16];
    const float* ffn_in_b  = (const float*)d_in[17];
    const float* ffn_dw_w  = (const float*)d_in[18];
    const float* ffn_dw_b  = (const float*)d_in[19];
    const float* ffn_fft   = (const float*)d_in[20];
    const float* ffn_out_w = (const float*)d_in[21];
    const float* ffn_out_b = (const float*)d_in[22];

    // workspace carve-up
    const size_t SZ_F32 = (size_t)NPIX * DIM * sizeof(float);          // 48 MB
    const size_t SZ_T   = (size_t)NPIX * 384 * sizeof(bf16_t);         // 192 MB
    const size_t SZ_D   = (size_t)NPIX * 288 * sizeof(bf16_t);         // 144 MB
    char* ws = (char*)d_ws;
    float*  x1   = (float*)ws;                                 // also reused for h (FFN out)
    float*  x2   = (float*)(ws + SZ_F32);
    bf16_t* tbuf = (bf16_t*)(ws + 2 * SZ_F32);                 // conv1x1 outputs (<=384ch)
    bf16_t* dbuf = (bf16_t*)(ws + 2 * SZ_F32 + SZ_T);          // dwconv / gated outputs
    float*  wker = (float*)(ws + 2 * SZ_F32 + SZ_T + SZ_D);    // 48*64 filter kernels

    const int MTILES = NPIX / 64;  // 4096 GEMM blocks

    // K0: spectral filter -> spatial kernels
    build_ffn_filter_kernel<<<(DIM * 64 + 255) / 256, 256, 0, stream>>>(ffn_fft, wker);

    // K1: x1 = x + pos_dwconv7(x)
    posconv_kernel<<<(BATCH * DIM * HWSZ) / 256, 256, 0, stream>>>(x, pos_w, pos_b, x1);

    // K2: tbuf = conv1x1(bn1(x1)) 48 -> 288 (bf16, WMMA)
    {
        size_t smem = (size_t)(64 * 64 + 64 * 288) * sizeof(bf16_t);   // 44 KB
        conv1x1_wmma_kernel<true, false, true, false, 64, 288><<<MTILES, 256, smem, stream>>>(
            x1, bn1_w, bn1_b, bn1_m, bn1_v, efa_hid_w, nullptr, tbuf, DIM);
    }

    // K3: dbuf = dwconv3(tbuf), 288 ch, no bias
    dwconv3_bf16_kernel<<<((long long)BATCH * 288 * HWSZ) / 256, 256, 0, stream>>>(
        tbuf, efa_dw_w, dbuf, 288);

    // K4: EFA patch core -> x2 = x1 + efa(...)
    efa_patch_kernel<<<BATCH * (WW / 8) * (HH / 8), 256, 0, stream>>>(
        dbuf, efa_ln_w, efa_ln_b, efa_out_w, x1, x2);

    // K5: tbuf = conv1x1(bn2(x2)) + ffn_in_b, 48 -> 384 (bf16, WMMA)
    {
        size_t smem = (size_t)(64 * 64 + 64 * 384) * sizeof(bf16_t);   // 56 KB
        conv1x1_wmma_kernel<true, false, true, true, 64, 384><<<MTILES, 256, smem, stream>>>(
            x2, bn2_w, bn2_b, bn2_m, bn2_v, ffn_in_w, ffn_in_b, tbuf, DIM);
    }

    // K6: dbuf = gelu(dw3(tbuf)[0:192]) * dw3(tbuf)[192:384]
    dwconv3_gate_kernel<<<((long long)BATCH * 192 * HWSZ) / 256, 256, 0, stream>>>(
        tbuf, ffn_dw_w, ffn_dw_b, dbuf, 192);

    // K7: h(x1 region, fp32) = conv1x1(dbuf) 192 -> 48 + ffn_out_b (WMMA)
    {
        size_t smem = (size_t)(64 * 192 + 192 * 48) * sizeof(bf16_t);  // 42 KB
        conv1x1_wmma_kernel<false, true, false, true, 192, 48><<<MTILES, 256, smem, stream>>>(
            dbuf, nullptr, nullptr, nullptr, nullptr, ffn_out_w, ffn_out_b, x1, 192);
    }

    // K8: out = x2 + per-channel 8x8 circular filter of h
    ffn_spectral_kernel<<<BATCH * DIM * ((WW / 8) * (HH / 8) / 4), 256, 0, stream>>>(
        x1, wker, x2, (float*)d_out);
}
